// ScaledDotAttention_15745350107334
// MI455X (gfx1250) — matrix operand order repaired and checked by reading, not compiled, not run on hardware
//
#include <hip/hip_runtime.h>
#include <hip/hip_bf16.h>

typedef __bf16 bf16_t;
typedef __attribute__((ext_vector_type(16))) bf16_t bf16x16;
typedef __attribute__((ext_vector_type(8)))  bf16_t bf16x8;
typedef __attribute__((ext_vector_type(4)))  bf16_t bf16x4;
typedef __attribute__((ext_vector_type(8)))  float  v8f;
typedef __attribute__((ext_vector_type(4)))  unsigned int u32x4;
typedef __attribute__((ext_vector_type(8)))  int i32x8;
typedef __attribute__((ext_vector_type(4)))  int i32x4;

#define B_   8
#define T_   2048
#define D_   512

// Dynamic LDS layout for the attention kernel (82944 B)
#define KT0_OFF   0        // Kt tile buffer 0 : 512 x 32 bf16 = 32768 B
#define KT1_OFF   32768    // Kt tile buffer 1 : 32768 B
#define PART_OFF  65536    // 8 waves * (16x32 f32 partial) = 16384 B
#define P_OFF     81920    // P[16][32] bf16 = 1024 B
#define SMEM_BYTES 82944

union F8 { v8f v; float f[8]; float4 q[2]; };

__device__ __forceinline__ bf16x16 cvt_bf16x16(float4 a, float4 b, float4 c, float4 d) {
  bf16x16 r;
  r[0]=(bf16_t)a.x;  r[1]=(bf16_t)a.y;  r[2]=(bf16_t)a.z;  r[3]=(bf16_t)a.w;
  r[4]=(bf16_t)b.x;  r[5]=(bf16_t)b.y;  r[6]=(bf16_t)b.z;  r[7]=(bf16_t)b.w;
  r[8]=(bf16_t)c.x;  r[9]=(bf16_t)c.y;  r[10]=(bf16_t)c.z; r[11]=(bf16_t)c.w;
  r[12]=(bf16_t)d.x; r[13]=(bf16_t)d.y; r[14]=(bf16_t)d.z; r[15]=(bf16_t)d.w;
  return r;
}

// TDM: DMA one 2D tile (32 keys x 512 d, bf16) of Kt[b][d][t] into LDS.
// D# per CDNA5 ISA 8.3/8.4: group0 = {count/flags, lds_addr, global_addr lo,
// global_addr hi | type=2}; group1 = dims/strides in data_size units.
__device__ __forceinline__ void tdm_load_ktile(unsigned lds_off, const bf16_t* gsrc) {
  unsigned long long ga = (unsigned long long)(uintptr_t)gsrc;
  u32x4 g0;
  g0[0] = 1u;                                          // count=1, user mode
  g0[1] = lds_off;                                     // LDS byte address
  g0[2] = (unsigned)(ga & 0xffffffffu);                // global_addr[31:0]
  g0[3] = (unsigned)((ga >> 32) & 0x01ffffffu)         // global_addr[56:32]
        | (2u << 30);                                  // type=2 ("image")
  i32x8 g1;
  g1[0] = (int)(1u << 16);          // workgroup_mask=0, data_size=1 (2 bytes)
  g1[1] = (int)(2048u << 16);       // tensor_dim0[15:0]=2048 (keys, contiguous dim)
  g1[2] = (int)(512u  << 16);       // tensor_dim0 hi=0 | tensor_dim1[15:0]=512 (d)
  g1[3] = (int)(32u   << 16);       // tensor_dim1 hi=0 | tile_dim0=32 keys
  g1[4] = 512;                      // tile_dim1=512, tile_dim2=0
  g1[5] = 2048;                     // tensor_dim0_stride=2048 elements
  g1[6] = 0;                        // stride hi | tensor_dim1_stride lo
  g1[7] = 0;
  i32x4 z4 = {0, 0, 0, 0};          // 2D tensor: groups 2/3 unused
#if __clang_major__ >= 23
  i32x8 z8 = {0, 0, 0, 0, 0, 0, 0, 0};
  __builtin_amdgcn_tensor_load_to_lds(g0, g1, z4, z4, z8, 0);
#else
  __builtin_amdgcn_tensor_load_to_lds(g0, g1, z4, z4, 0);
#endif
}

// ---- prep 1: K (f32 [b][t][d]) -> Kt (bf16 [b][d][t]) via LDS tile transpose ----
__global__ __launch_bounds__(256) void k_transpose_bf16(const float* __restrict__ K,
                                                        bf16_t* __restrict__ Kt) {
  __shared__ bf16_t tile[64][65];   // padded: no bank conflicts
  const int tid = threadIdx.x;
  const int blk = blockIdx.x;
  const int b   = blk >> 8;                  // / (32*8)
  const int tb  = ((blk >> 3) & 31) << 6;    // t tile base
  const int db  = (blk & 7) << 6;            // d tile base
  #pragma unroll
  for (int i = 0; i < 16; ++i) {
    int e  = tid + i * 256;          // [0,4096)
    int tt = e >> 6, dd = e & 63;    // consecutive tid -> consecutive d (coalesced)
    tile[dd][tt] = (bf16_t)K[((size_t)b * T_ + tb + tt) * D_ + db + dd];
  }
  __syncthreads();
  #pragma unroll
  for (int i = 0; i < 16; ++i) {
    int e  = tid + i * 256;
    int dd = e >> 6, tt = e & 63;    // consecutive tid -> consecutive t (coalesced)
    Kt[((size_t)b * D_ + db + dd) * T_ + tb + tt] = tile[dd][tt];
  }
}

// ---- prep 2: V (f32) -> bf16, same layout ----
__global__ __launch_bounds__(256) void v_convert_bf16(const float* __restrict__ V,
                                                      bf16_t* __restrict__ Vb) {
  size_t i = ((size_t)blockIdx.x * 256 + threadIdx.x) * 4;
  float4 v = *(const float4*)(V + i);
  bf16x4 o;
  o[0] = (bf16_t)v.x; o[1] = (bf16_t)v.y; o[2] = (bf16_t)v.z; o[3] = (bf16_t)v.w;
  *(bf16x4*)(Vb + i) = o;
}

// ---- main attention kernel ----
__global__ __launch_bounds__(256)
void sdpa_flash_bf16_wmma(const float*  __restrict__ Qp,
                          const bf16_t* __restrict__ Ktg,   // [b][d][t] bf16
                          const bf16_t* __restrict__ Vg,    // [b][t][d] bf16
                          const int*    __restrict__ mfp,
                          float*        __restrict__ Op)
{
  extern __shared__ __align__(16) char smem[];
  const int tid  = threadIdx.x;
  const int wave = tid >> 5;        // 0..7 : owns D columns [wave*64, wave*64+64)
  const int lane = tid & 31;
  const int col  = lane & 15;       // C-layout column / A-layout row
  const int half = lane >> 4;
  const int wg   = blockIdx.x;
  const int b    = wg >> 7;
  const int qbase= (wg & 127) << 4;
  const int mf   = mfp[0];
  const float scale = 0.044194173824159216f; // 1/sqrt(512)

  bf16_t* sP = (bf16_t*)(smem + P_OFF);
  const bf16_t* KtB = Ktg + (size_t)b * D_ * T_;

  // Q A-fragments (bf16, exact ISA A-layout), resident for the whole kernel
  bf16x16 aQ[2];
  {
    const float* qrow = Qp + ((size_t)b * T_ + qbase + col) * D_;
    #pragma unroll
    for (int f = 0; f < 2; ++f) {
      const float* qp = qrow + wave * 64 + f * 32 + half * 8;
      float4 q0 = *(const float4*)(qp + 0);
      float4 q1 = *(const float4*)(qp + 4);
      float4 q2 = *(const float4*)(qp + 16);
      float4 q3 = *(const float4*)(qp + 20);
      aQ[f] = cvt_bf16x16(q0, q1, q2, q3);
    }
  }

  F8 acc[4];
  #pragma unroll
  for (int c = 0; c < 4; ++c)
    #pragma unroll
    for (int i = 0; i < 8; ++i) acc[c].f[i] = 0.0f;
  float mrow[8], lrow[8];
  #pragma unroll
  for (int r = 0; r < 8; ++r) { mrow[r] = -__builtin_inff(); lrow[r] = 0.0f; }

  // TDM prologue: DMA first K tile into buffer 0
  if (wave == 0) tdm_load_ktile((unsigned)(uintptr_t)(smem + KT0_OFF), KtB);

  for (int kbi = 0; kbi < T_ / 32; ++kbi) {
    const int kb = kbi * 32;
    bf16_t* sKt = (bf16_t*)(smem + ((kbi & 1) ? KT1_OFF : KT0_OFF));

    if (wave == 0) __builtin_amdgcn_s_wait_tensorcnt(0);  // current tile landed
    __syncthreads();
    if (wave == 0 && kbi + 1 < T_ / 32) {                 // overlap next DMA
      unsigned nxt = (unsigned)(uintptr_t)(smem + ((kbi & 1) ? KT0_OFF : KT1_OFF));
      tdm_load_ktile(nxt, KtB + kb + 32);
    }
    if (kbi + 1 < T_ / 32)  // warm L2->WGP path for next V block
      __builtin_prefetch(Vg + ((size_t)b * T_ + kb + 32) * D_ + tid * 64, 0, 0);

    // ---- partial S = Q[:,dslice] x Kt[dslice,:] : 4 WMMAs per wave ----
    F8 s0, s1;
    #pragma unroll
    for (int i = 0; i < 8; ++i) { s0.f[i] = 0.0f; s1.f[i] = 0.0f; }
    #pragma unroll
    for (int f = 0; f < 2; ++f) {
      const int dbase = wave * 64 + f * 32;
      bf16x16 bK0, bK1;
      #pragma unroll
      for (int i = 0; i < 16; ++i) {
        const int kk = (i < 8) ? (half * 8 + i) : (16 + half * 8 + (i - 8));
        bK0[i] = sKt[(dbase + kk) * 32 + col];
        bK1[i] = sKt[(dbase + kk) * 32 + 16 + col];
      }
      s0.v = __builtin_amdgcn_wmma_f32_16x16x32_bf16(false, aQ[f], false, bK0,
                                                     (short)0, s0.v, false, false);
      s1.v = __builtin_amdgcn_wmma_f32_16x16x32_bf16(false, aQ[f], false, bK1,
                                                     (short)0, s1.v, false, false);
    }
    {
      float4* pp = (float4*)(smem + PART_OFF + wave * 2048 + lane * 64);
      pp[0] = s0.q[0]; pp[1] = s0.q[1]; pp[2] = s1.q[0]; pp[3] = s1.q[1];
    }
    __syncthreads();

    // ---- reduce partials over the 8 waves (identical per-lane layout) ----
    F8 S0, S1;
    #pragma unroll
    for (int i = 0; i < 8; ++i) { S0.f[i] = 0.0f; S1.f[i] = 0.0f; }
    #pragma unroll
    for (int ww = 0; ww < 8; ++ww) {
      const v8f* rp = (const v8f*)(smem + PART_OFF + ww * 2048 + lane * 64);
      S0.v += rp[0];
      S1.v += rp[1];
    }

    // ---- multiplicative mask, scale, online softmax ----
    float p0[8], p1[8];
    #pragma unroll
    for (int r = 0; r < 8; ++r) {
      const int trow = qbase + half * 8 + r;
      float x0 = S0.f[r] * ((mf && (kb + col)      >= trow) ? -1e9f : 1.0f) * scale;
      float x1 = S1.f[r] * ((mf && (kb + 16 + col) >= trow) ? -1e9f : 1.0f) * scale;
      float mx = fmaxf(x0, x1);
      #pragma unroll
      for (int msk = 1; msk < 16; msk <<= 1)
        mx = fmaxf(mx, __shfl_xor(mx, msk, 32));
      float mnew  = fmaxf(mrow[r], mx);
      float alpha = __expf(mrow[r] - mnew);
      float e0 = __expf(x0 - mnew);
      float e1 = __expf(x1 - mnew);
      float ls = e0 + e1;
      #pragma unroll
      for (int msk = 1; msk < 16; msk <<= 1)
        ls += __shfl_xor(ls, msk, 32);
      lrow[r] = lrow[r] * alpha + ls;
      mrow[r] = mnew;
      #pragma unroll
      for (int c = 0; c < 4; ++c) acc[c].f[r] *= alpha;
      p0[r] = e0; p1[r] = e1;
    }

    // ---- wave 0 publishes P (16x32 bf16 row-major) ----
    if (wave == 0) {
      #pragma unroll
      for (int r = 0; r < 8; ++r) {
        int m = half * 8 + r;
        sP[m * 32 + col]      = (bf16_t)p0[r];
        sP[m * 32 + 16 + col] = (bf16_t)p1[r];
      }
    }
    __syncthreads();

    // ---- PV: acc[c] += P(16x32) x V(32x16); V is pre-converted bf16 (L2) ----
    bf16x16 aP;
    {
      const bf16_t* pr = sP + col * 32 + half * 8;
      bf16x8 lo = *(const bf16x8*)(pr);
      bf16x8 hi = *(const bf16x8*)(pr + 16);
      #pragma unroll
      for (int i = 0; i < 8; ++i) { aP[i] = lo[i]; aP[i + 8] = hi[i]; }
    }
    const bf16_t* Vb = Vg + ((size_t)b * T_ + kb) * D_ + wave * 64;
    #pragma unroll
    for (int c = 0; c < 4; ++c) {
      bf16x16 bV;
      #pragma unroll
      for (int i = 0; i < 16; ++i) {
        const int kk = (i < 8) ? (half * 8 + i) : (16 + half * 8 + (i - 8));
        bV[i] = Vb[(size_t)kk * D_ + c * 16 + col];
      }
      acc[c].v = __builtin_amdgcn_wmma_f32_16x16x32_bf16(false, aP, false, bV,
                                                         (short)0, acc[c].v, false, false);
    }
    __syncthreads();   // sP/sKt readers done before next iteration's writers
  }

  // ---- epilogue: O = acc / l ----
  float* Ob = Op + ((size_t)b * T_ + qbase) * D_ + wave * 64;
  #pragma unroll
  for (int r = 0; r < 8; ++r) {
    float inv = 1.0f / lrow[r];
    int m = half * 8 + r;
    #pragma unroll
    for (int c = 0; c < 4; ++c)
      Ob[(size_t)m * D_ + c * 16 + col] = acc[c].f[r] * inv;
  }
}

extern "C" void kernel_launch(void* const* d_in, const int* in_sizes, int n_in,
                              void* d_out, int out_size, void* d_ws, size_t ws_size,
                              hipStream_t stream) {
  const float* Q  = (const float*)d_in[0];
  const float* K  = (const float*)d_in[1];
  const float* V  = (const float*)d_in[2];
  const int*   mf = (const int*)d_in[3];
  float* out = (float*)d_out;
  (void)in_sizes; (void)n_in; (void)out_size; (void)ws_size;

  // workspace: Kt bf16 (16 MB) | V bf16 (16 MB)
  bf16_t* Kt = (bf16_t*)d_ws;
  bf16_t* Vb = (bf16_t*)((char*)d_ws + (size_t)B_ * D_ * T_ * sizeof(bf16_t));

  k_transpose_bf16<<<dim3(B_ * (T_ / 64) * (D_ / 64)), dim3(256), 0, stream>>>(K, Kt);
  v_convert_bf16 <<<dim3((B_ * T_ * D_) / (4 * 256)), dim3(256), 0, stream>>>(V, Vb);
  sdpa_flash_bf16_wmma<<<dim3(B_ * (T_ / 16)), dim3(256), SMEM_BYTES, stream>>>(
      Q, Kt, Vb, mf, out);
}
